// _BaseAttention_65481071398925
// MI455X (gfx1250) — matrix operand order repaired and checked by reading, not compiled, not run on hardware
//
#include <hip/hip_runtime.h>
#include <hip/hip_bf16.h>

// ---------------------------------------------------------------------------
// Causal multi-head attention for MI455X (gfx1250).
// B=2, S=2048, EMBED=1024, H=16, D_K=64.  Outputs: out [B,S,E] f32 then
// attn [B,H,S,S] f32, concatenated in d_out.
// Bandwidth-bound on the 512MB attn write -> bf16 WMMA matmuls, full 32-row
// score strip resident in LDS (256KB+pad < 320KB/WGP), NT stores for the
// streaming attn/out writes, global_prefetch for next K/V tiles.
// ---------------------------------------------------------------------------

typedef __bf16 bf16_t;
typedef __attribute__((ext_vector_type(16))) __bf16 v16bf;
typedef __attribute__((ext_vector_type(8)))  __bf16 v8bf;
typedef __attribute__((ext_vector_type(8)))  float  v8f;
typedef __attribute__((ext_vector_type(4)))  float  v4f;

#define BATCH 2
#define HEADS 16
#define SEQ   2048
#define DK    64
#define EMB   1024
#define QB    32          // query rows per workgroup
#define SP    2052        // LDS score-strip pitch in floats (2048 + 4 pad)

// ---------------------------------------------------------------------------
// Repack: Q -> bf16 [b,h,s,d] pre-scaled by 1/sqrt(dk); K -> bf16 [b,h,s,d];
// V -> bf16 transposed [b,h,d,s] so PV B-fragments are contiguous.
// ---------------------------------------------------------------------------
__global__ void cvt_qkv_kernel(const float* __restrict__ q,
                               const float* __restrict__ k,
                               const float* __restrict__ v,
                               bf16_t* __restrict__ qb,
                               bf16_t* __restrict__ kb,
                               bf16_t* __restrict__ vt) {
    int idx = blockIdx.x * blockDim.x + threadIdx.x;   // over B*S*EMB
    int d =  idx        & (DK - 1);
    int h = (idx >> 6)  & (HEADS - 1);
    int s = (idx >> 10) & (SEQ - 1);
    int b =  idx >> 21;
    int bh = b * HEADS + h;
    qb[((size_t)bh * SEQ + s) * DK + d] = (bf16_t)(q[idx] * 0.125f); // 1/sqrt(64)
    kb[((size_t)bh * SEQ + s) * DK + d] = (bf16_t)(k[idx]);
    vt[((size_t)bh * DK  + d) * SEQ + s] = (bf16_t)(v[idx]);
}

// ---------------------------------------------------------------------------
// Fused attention. grid = (SEQ/QB, HEADS, BATCH), block = 256 (8 wave32).
// Dynamic LDS: QB * SP * 4 bytes = 262656 B.
// ---------------------------------------------------------------------------
__global__ void __launch_bounds__(256)
attn_kernel(const bf16_t* __restrict__ qb,
            const bf16_t* __restrict__ kb,
            const bf16_t* __restrict__ vt,
            float* __restrict__ out,
            float* __restrict__ attn) {
    extern __shared__ float P[];                 // [QB][SP] score strip

    const int q0   = blockIdx.x * QB;
    const int h    = blockIdx.y;
    const int b    = blockIdx.z;
    const int bh   = b * HEADS + h;
    const int tid  = threadIdx.x;
    const int w    = tid >> 5;                   // wave id 0..7
    const int lane = tid & 31;
    const int lo   = lane & 15;
    const int hi   = lane >> 4;
    const int Kc   = q0 + QB;                    // causal key extent (mult of 32)

    // ---------------- Phase 1: S = (Q/sqrt(dk)) * K^T into LDS -------------
    {
        const int qt = w & 1;                    // which 16-row q tile
        // Q A-fragments for d-chunks [0,32) and [32,64): contiguous 16B loads.
        v16bf aq[2];
        const bf16_t* qrow = qb + ((size_t)(bh * SEQ + q0 + qt * 16 + lo)) * DK;
#pragma unroll
        for (int c = 0; c < 2; ++c) {
            v8bf r0 = *(const v8bf*)(qrow + c * 32 + hi * 16);
            v8bf r1 = *(const v8bf*)(qrow + c * 32 + hi * 16 + 8);
#pragma unroll
            for (int i = 0; i < 8; ++i) { aq[c][i] = r0[i]; aq[c][8 + i] = r1[i]; }
        }
        const int nkt = Kc >> 4;                 // 16-key tiles needed
        for (int kt = (w >> 1); kt < nkt; kt += 4) {
            // K B-fragment: lane holds key row kt*16+lo, 16 consecutive d.
            const bf16_t* krow = kb + ((size_t)(bh * SEQ + kt * 16 + lo)) * DK;
            // Prefetch next tile this wave will touch (speculative, OOB-safe).
            __builtin_prefetch(krow + (size_t)64 * DK, 0, 3);
            v16bf bk0 = *(const v16bf*)(krow + hi * 16);
            v16bf bk1 = *(const v16bf*)(krow + 32 + hi * 16);
            v8f c8 = {};
            c8 = __builtin_amdgcn_wmma_f32_16x16x32_bf16(false, aq[0], false, bk0,
                                                         (short)0, c8, false, false);
            c8 = __builtin_amdgcn_wmma_f32_16x16x32_bf16(false, aq[1], false, bk1,
                                                         (short)0, c8, false, false);
            const int key = kt * 16 + lo;
#pragma unroll
            for (int r = 0; r < 8; ++r) {
                const int row = qt * 16 + r + hi * 8;     // row within block
                const int qg  = q0 + row;                 // global q index
                P[row * SP + key] = (key <= qg) ? c8[r] : -1.0e30f;
            }
        }
    }
    __syncthreads();

    // ---------------- Phase 2: softmax rows, write attn (NT stores) --------
    {
#pragma unroll
        for (int rr = 0; rr < 4; ++rr) {
            const int row = w * 4 + rr;
            const int qg  = q0 + row;
            float* Pr = P + row * SP;
            // row max (vectorized: 128 floats per wave iteration)
            float m = -1.0e30f;
            for (int k = lane * 4; k < Kc; k += 128) {
                v4f x = *(v4f*)(Pr + k);
                m = fmaxf(m, fmaxf(fmaxf(x[0], x[1]), fmaxf(x[2], x[3])));
            }
#pragma unroll
            for (int off = 16; off > 0; off >>= 1) m = fmaxf(m, __shfl_xor(m, off, 32));
            // exp + row sum (vectorized)
            float sum = 0.f;
            for (int k = lane * 4; k < Kc; k += 128) {
                v4f x = *(v4f*)(Pr + k);
                v4f p;
#pragma unroll
                for (int i = 0; i < 4; ++i) p[i] = __expf(x[i] - m);
                *(v4f*)(Pr + k) = p;
                sum += (p[0] + p[1]) + (p[2] + p[3]);
            }
#pragma unroll
            for (int off = 16; off > 0; off >>= 1) sum += __shfl_xor(sum, off, 32);
            const float inv = 1.0f / sum;
            // normalize in LDS + stream to global attn (non-temporal: attn is
            // write-once, never re-read; keep L2 for the reused K/V instead)
            float* arow = attn + ((size_t)(bh * SEQ + qg)) * SEQ;
            for (int k = lane * 4; k < Kc; k += 128) {
                v4f pv = *(v4f*)(Pr + k);
                pv *= inv;
                *(v4f*)(Pr + k) = pv;
                __builtin_nontemporal_store(pv, (v4f*)(arow + k));
            }
            v4f z = {};
            for (int k = Kc + lane * 4; k < SEQ; k += 128)
                __builtin_nontemporal_store(z, (v4f*)(arow + k));
        }
    }
    __syncthreads();

    // ---------------- Phase 3: out = P * V ---------------------------------
    {
        const int qt = w >> 2;                   // q row tile 0..1
        const int nc = w & 3;                    // d column tile 0..3
        const int nch = Kc >> 5;                 // 32-key chunks
        v8f c8 = {};
        const float*  Pr   = P  + (qt * 16 + lo) * SP;
        const bf16_t* vrow = vt + ((size_t)(bh * DK + nc * 16 + lo)) * SEQ;
        for (int c32 = 0; c32 < nch; ++c32) {
            // Prefetch next V chunk (speculative).
            __builtin_prefetch(vrow + (c32 + 1) * 32 + hi * 16, 0, 3);
            // A-frag: probs row lo, keys c32*32 + hi*16 + {0..15}
            const float* base = Pr + c32 * 32 + hi * 16;
            v4f f0 = *(const v4f*)(base);
            v4f f1 = *(const v4f*)(base + 4);
            v4f f2 = *(const v4f*)(base + 8);
            v4f f3 = *(const v4f*)(base + 12);
            v16bf ap;
#pragma unroll
            for (int i = 0; i < 4; ++i) {
                ap[i]      = (bf16_t)f0[i];
                ap[4 + i]  = (bf16_t)f1[i];
                ap[8 + i]  = (bf16_t)f2[i];
                ap[12 + i] = (bf16_t)f3[i];
            }
            // B-frag: V^T, lane holds d = nc*16+lo, 16 consecutive keys.
            v16bf bv = *(const v16bf*)(vrow + c32 * 32 + hi * 16);
            c8 = __builtin_amdgcn_wmma_f32_16x16x32_bf16(false, ap, false, bv,
                                                         (short)0, c8, false, false);
        }
#pragma unroll
        for (int r = 0; r < 8; ++r) {
            const int qg = q0 + qt * 16 + r + hi * 8;
            __builtin_nontemporal_store(
                c8[r], out + ((size_t)(b * SEQ + qg)) * EMB + h * DK + nc * 16 + lo);
        }
    }
}

// ---------------------------------------------------------------------------
extern "C" void kernel_launch(void* const* d_in, const int* in_sizes, int n_in,
                              void* d_out, int out_size, void* d_ws, size_t ws_size,
                              hipStream_t stream) {
    (void)in_sizes; (void)n_in; (void)out_size; (void)ws_size;
    const float* q = (const float*)d_in[0];
    const float* k = (const float*)d_in[1];
    const float* v = (const float*)d_in[2];
    // d_in[3] is the causal mask; causality is applied analytically.

    float* out  = (float*)d_out;
    float* attn = out + (size_t)BATCH * SEQ * EMB;

    const size_t nElem = (size_t)BATCH * HEADS * SEQ * DK;   // 4M bf16 each
    bf16_t* qb = (bf16_t*)d_ws;
    bf16_t* kb = qb + nElem;
    bf16_t* vt = kb + nElem;

    const int total = BATCH * SEQ * EMB;                      // 4,194,304
    cvt_qkv_kernel<<<total / 256, 256, 0, stream>>>(q, k, v, qb, kb, vt);

    dim3 grid(SEQ / QB, HEADS, BATCH);
    attn_kernel<<<grid, 256, QB * SP * sizeof(float), stream>>>(qb, kb, vt, out, attn);
}
